// encoder_69131793596858
// MI455X (gfx1250) — matrix operand order repaired and checked by reading, not compiled, not run on hardware
//
#include <hip/hip_runtime.h>
#include <hip/hip_bf16.h>

// ---------------------------------------------------------------------------
// GNN encoder forward for MI455X (gfx1250).
// All 128x128 linears run on v_wmma_f32_16x16x32_f16 (f32 accumulate).
// Memory-bound workload: gather/scatter passes fused, atomics for segment ops,
// wave32 shuffle reductions for per-row LayerNorm (D=128 -> 4 elems/lane).
// ---------------------------------------------------------------------------

typedef _Float16 v4h  __attribute__((ext_vector_type(4)));
typedef _Float16 v8h  __attribute__((ext_vector_type(8)));
typedef _Float16 v16h __attribute__((ext_vector_type(16)));
typedef float    v8f  __attribute__((ext_vector_type(8)));

#define DD   128
#define LDSP 136   // padded halfs per LDS row (16B-aligned, breaks bank cycles)

__device__ __forceinline__ float wave_sum(float v) {
#pragma unroll
  for (int off = 16; off > 0; off >>= 1) v += __shfl_xor(v, off, 32);
  return v;
}
__device__ __forceinline__ float sigf(float x) { return 1.f / (1.f + __expf(-x)); }
__device__ __forceinline__ float siluf(float x) { return x * sigf(x); }

// --- weight convert: W[k][n] f32  ->  Wt[n][k] f16 (column-major for B frag) --
__global__ void k_wconv(const float* __restrict__ w, _Float16* __restrict__ wt) {
  int idx = blockIdx.x * 256 + threadIdx.x;      // 16384 elements
  int n = idx >> 7, k = idx & 127;
  wt[idx] = (_Float16)w[k * DD + n];
}

// --- GEMM: Y[M,128] = act(X[M,128] @ W + b), WMMA f16->f32 ------------------
// Block: 256 thr = 8 waves. 64-row tile. wave w: rowstrip = w&3, n-half = w>>2.
template <int ACT>
__global__ __launch_bounds__(256) void k_gemm(const float* __restrict__ X,
                                              const _Float16* __restrict__ Wt,
                                              const float* __restrict__ bias,
                                              float* __restrict__ Y, int M) {
  __shared__ __align__(16) _Float16 ldsX[64 * LDSP];
  __shared__ __align__(16) _Float16 ldsW[128 * LDSP];
  __shared__ float ldsB[128];

  const int tid  = threadIdx.x;
  const int lane = tid & 31;
  const int wv   = tid >> 5;
  const int m0   = blockIdx.x * 64;

  { // stage W^T (f16): 2 threads per n-row, 64 halfs (128B) each
    const int r  = tid >> 1;
    const int hh = (tid & 1) * 64;
    const uint4* src = (const uint4*)(Wt + r * DD + hh);
    uint4* dst = (uint4*)(&ldsW[r * LDSP + hh]);
#pragma unroll
    for (int i = 0; i < 8; ++i) dst[i] = src[i];
  }
  if (tid < 128) ldsB[tid] = bias[tid];
  { // stage X tile, f32 -> f16: 4 threads per row, 32 cols each
    const int r  = tid >> 2;
    const int c0 = (tid & 3) * 32;
    const int m  = m0 + r;
    if (m < M) {
      const float4* src = (const float4*)(X + (size_t)m * DD + c0);
#pragma unroll
      for (int i = 0; i < 8; ++i) {
        float4 v = src[i];
        v4h h = {(_Float16)v.x, (_Float16)v.y, (_Float16)v.z, (_Float16)v.w};
        *(v4h*)(&ldsX[r * LDSP + c0 + i * 4]) = h;
      }
    } else {
      v4h z = {};
#pragma unroll
      for (int i = 0; i < 8; ++i) *(v4h*)(&ldsX[r * LDSP + c0 + i * 4]) = z;
    }
  }
  __syncthreads();

  const int rs   = wv & 3;           // row strip 0..3
  const int nh   = wv >> 2;          // n half 0..1
  const int arow = rs * 16 + (lane & 15);
  const int kg   = lane >> 4;        // K-group per ISA A-layout

  v8f acc[4] = {};

#pragma unroll
  for (int kk = 0; kk < 4; ++kk) {
    const int k0 = kk * 32;
    v16h a;  // A 16x32 f16: lanes 0-15 K {0..7,16..23}, lanes 16-31 +8
    {
      const v8h lo = *(const v8h*)(&ldsX[arow * LDSP + k0 + kg * 8]);
      const v8h hi = *(const v8h*)(&ldsX[arow * LDSP + k0 + 16 + kg * 8]);
#pragma unroll
      for (int i = 0; i < 8; ++i) { a[i] = lo[i]; a[8 + i] = hi[i]; }
    }
#pragma unroll
    for (int nt = 0; nt < 4; ++nt) {
      const int ncol = (nh * 4 + nt) * 16 + (lane & 15);
      v16h b;  // B 32x16: lane holds col n, lanes 0-15 K {0..7,16..23}, lanes 16-31 +8
      const v8h b0 = *(const v8h*)(&ldsW[ncol * LDSP + k0 + kg * 8]);
      const v8h b1 = *(const v8h*)(&ldsW[ncol * LDSP + k0 + 16 + kg * 8]);
#pragma unroll
      for (int i = 0; i < 8; ++i) { b[i] = b0[i]; b[8 + i] = b1[i]; }
      acc[nt] = __builtin_amdgcn_wmma_f32_16x16x32_f16(
          false, a, false, b, (short)0, acc[nt], false, false);
    }
  }

  const int nlane = lane & 15;
  const int moff  = (lane >> 4) * 8;
#pragma unroll
  for (int nt = 0; nt < 4; ++nt) {
    const int nn = (nh * 4 + nt) * 16 + nlane;
    const float bb = ldsB[nn];
#pragma unroll
    for (int v = 0; v < 8; ++v) {
      const int m = m0 + rs * 16 + moff + v;
      if (m < M) {
        float o = acc[nt][v] + bb;
        if (ACT == 1) o = fmaxf(o, 0.f);
        Y[(size_t)m * DD + nn] = o;
      }
    }
  }
}

// --- eggc edge pass: e = Ax[s]+Bx[d]+Cy ; sig-gated message + segment sums ---
__global__ void k_eggc_edge(const float* __restrict__ Ax, const float* __restrict__ Bx,
                            const float* __restrict__ Wsx, float* __restrict__ e_io,
                            float* __restrict__ num, float* __restrict__ den,
                            const int* __restrict__ src, const int* __restrict__ dst,
                            int E) {
  size_t t = (size_t)blockIdx.x * 256 + threadIdx.x;
  if (t >= (size_t)E * DD) return;
  int i = (int)(t >> 7), d = (int)(t & 127);
  int s = src[i], dn = dst[i];
  float e  = Ax[(size_t)s * DD + d] + Bx[(size_t)dn * DD + d] + e_io[t];
  float sg = sigf(e);
  e_io[t]  = e;
  atomicAdd(&num[(size_t)dn * DD + d], sg * Wsx[(size_t)s * DD + d]);
  atomicAdd(&den[(size_t)dn * DD + d], sg);
}

// --- eggc node epilogue: x += silu(LN(Wdx + num/(den+1e-6))) ----------------
__global__ __launch_bounds__(256) void k_eggc_node(float* __restrict__ x_io,
    const float* __restrict__ Wdx, const float* __restrict__ num,
    const float* __restrict__ den, const float* __restrict__ g,
    const float* __restrict__ b, int Nr) {
  int row = blockIdx.x * 8 + (threadIdx.x >> 5);
  if (row >= Nr) return;
  int lane = threadIdx.x & 31;
  size_t base = (size_t)row * DD + lane * 4;
  float4 wd = *(const float4*)(Wdx + base);
  float4 nu = *(const float4*)(num + base);
  float4 de = *(const float4*)(den + base);
  float h0 = wd.x + nu.x / (de.x + 1e-6f), h1 = wd.y + nu.y / (de.y + 1e-6f);
  float h2 = wd.z + nu.z / (de.z + 1e-6f), h3 = wd.w + nu.w / (de.w + 1e-6f);
  float mean = wave_sum(h0 + h1 + h2 + h3) * 0.0078125f;
  float var  = wave_sum(h0*h0 + h1*h1 + h2*h2 + h3*h3) * 0.0078125f - mean * mean;
  float r    = rsqrtf(var + 1e-5f);
  float4 gv = *(const float4*)(g + lane * 4);
  float4 bv = *(const float4*)(b + lane * 4);
  float4 x  = *(float4*)(x_io + base);
  x.x += siluf((h0 - mean) * r * gv.x + bv.x);
  x.y += siluf((h1 - mean) * r * gv.y + bv.y);
  x.z += siluf((h2 - mean) * r * gv.z + bv.z);
  x.w += siluf((h3 - mean) * r * gv.w + bv.w);
  *(float4*)(x_io + base) = x;
}

// --- y += silu(LN(e)) -------------------------------------------------------
__global__ __launch_bounds__(256) void k_ln_silu_add(float* __restrict__ y_io,
    const float* __restrict__ e, const float* __restrict__ g,
    const float* __restrict__ b, int Nr) {
  int row = blockIdx.x * 8 + (threadIdx.x >> 5);
  if (row >= Nr) return;
  int lane = threadIdx.x & 31;
  size_t base = (size_t)row * DD + lane * 4;
  float4 t = *(const float4*)(e + base);
  float mean = wave_sum(t.x + t.y + t.z + t.w) * 0.0078125f;
  float var  = wave_sum(t.x*t.x + t.y*t.y + t.z*t.z + t.w*t.w) * 0.0078125f - mean*mean;
  float r    = rsqrtf(var + 1e-5f);
  float4 gv = *(const float4*)(g + lane * 4);
  float4 bv = *(const float4*)(b + lane * 4);
  float4 y  = *(float4*)(y_io + base);
  y.x += siluf((t.x - mean) * r * gv.x + bv.x);
  y.y += siluf((t.y - mean) * r * gv.y + bv.y);
  y.z += siluf((t.z - mean) * r * gv.z + bv.z);
  y.w += siluf((t.w - mean) * r * gv.w + bv.w);
  *(float4*)(y_io + base) = y;
}

// --- dst = res + LN(t) ------------------------------------------------------
__global__ __launch_bounds__(256) void k_ln_add(float* __restrict__ dst,
    const float* __restrict__ res, const float* __restrict__ t,
    const float* __restrict__ g, const float* __restrict__ b, int Nr) {
  int row = blockIdx.x * 8 + (threadIdx.x >> 5);
  if (row >= Nr) return;
  int lane = threadIdx.x & 31;
  size_t base = (size_t)row * DD + lane * 4;
  float4 tv = *(const float4*)(t + base);
  float mean = wave_sum(tv.x + tv.y + tv.z + tv.w) * 0.0078125f;
  float var  = wave_sum(tv.x*tv.x + tv.y*tv.y + tv.z*tv.z + tv.w*tv.w) * 0.0078125f - mean*mean;
  float r    = rsqrtf(var + 1e-5f);
  float4 gv = *(const float4*)(g + lane * 4);
  float4 bv = *(const float4*)(b + lane * 4);
  float4 rv = *(const float4*)(res + base);
  float4 o;
  o.x = rv.x + (tv.x - mean) * r * gv.x + bv.x;
  o.y = rv.y + (tv.y - mean) * r * gv.y + bv.y;
  o.z = rv.z + (tv.z - mean) * r * gv.z + bv.z;
  o.w = rv.w + (tv.w - mean) * r * gv.w + bv.w;
  *(float4*)(dst + base) = o;
}

// --- MHA pass 1: s = q[d]*k[s]*e/sqrt(32); store s, logit, segment-max ------
__global__ __launch_bounds__(256) void k_mha_edge1(const float* __restrict__ Q,
    const float* __restrict__ K, float* __restrict__ S_io, float* __restrict__ Lg,
    unsigned* __restrict__ Lmax, const int* __restrict__ src,
    const int* __restrict__ dst, int E) {
  int gid = blockIdx.x * 8 + (threadIdx.x >> 5);   // one wave per (edge, head)
  if (gid >= E * 4) return;
  int i = gid >> 2, h = gid & 3;
  int lane = threadIdx.x & 31;
  int d = h * 32 + lane;
  int s = src[i], dn = dst[i];
  float sv = Q[(size_t)dn * DD + d] * K[(size_t)s * DD + d] *
             S_io[(size_t)i * DD + d] * 0.17677669529663687f;  // 1/sqrt(32)
  S_io[(size_t)i * DD + d] = sv;
  float logit = wave_sum(sv);
  if (lane == 0) {
    Lg[(size_t)i * 4 + h] = logit;
    unsigned bits = __float_as_uint(logit);
    unsigned key = (bits & 0x80000000u) ? ~bits : (bits | 0x80000000u);
    atomicMax(&Lmax[(size_t)dn * 4 + h], key);   // monotone order-preserving key
  }
}

// --- MHA pass 2: a = exp(logit - max); num += a*v[src]; den += a ------------
__global__ void k_mha_edge2(const float* __restrict__ V, const float* __restrict__ Lg,
                            const unsigned* __restrict__ Lmax, float* __restrict__ num,
                            float* __restrict__ den, const int* __restrict__ src,
                            const int* __restrict__ dst, int E) {
  size_t t = (size_t)blockIdx.x * 256 + threadIdx.x;
  if (t >= (size_t)E * DD) return;
  int i = (int)(t >> 7), d = (int)(t & 127), h = d >> 5;
  int s = src[i], dn = dst[i];
  unsigned key = Lmax[(size_t)dn * 4 + h];
  float mx = (key & 0x80000000u) ? __uint_as_float(key ^ 0x80000000u)
                                 : __uint_as_float(~key);
  float a = __expf(Lg[(size_t)i * 4 + h] - mx);
  atomicAdd(&num[(size_t)dn * DD + d], a * V[(size_t)s * DD + d]);
  if ((d & 31) == 0) atomicAdd(&den[(size_t)dn * 4 + h], a);
}

// --- MHA combine: h = num / (den + 1e-6) ------------------------------------
__global__ void k_mha_hcomb(const float* __restrict__ num, const float* __restrict__ den,
                            float* __restrict__ hb, int Nn) {
  size_t t = (size_t)blockIdx.x * 256 + threadIdx.x;
  if (t >= (size_t)Nn * DD) return;
  int n = (int)(t >> 7), h = (int)((t & 127) >> 5);
  hb[t] = num[t] / (den[(size_t)n * 4 + h] + 1e-6f);
}

// ---------------------------------------------------------------------------
struct LinP { const float* b; const float* w; _Float16* wt; };
struct LnP  { const float* b; const float* g; };
struct EggcP { LinP A, B, C, Wd, Ws; LnP ln_e, ln_n; };

extern "C" void kernel_launch(void* const* d_in, const int* in_sizes, int n_in,
                              void* d_out, int out_size, void* d_ws, size_t ws_size,
                              hipStream_t stream) {
  (void)in_sizes; (void)out_size; (void)ws_size;
  const int N = 20000, EG = 240000, T = 400000, EF = 400000;
  if (n_in < 200) return;  // expects 10 arrays + 190 flattened param leaves

  const float* in_x = (const float*)d_in[0];
  const float* in_y = (const float*)d_in[1];
  const float* in_z = (const float*)d_in[2];
  const float* in_f = (const float*)d_in[3];
  const int* g_src  = (const int*)d_in[4];
  const int* g_dst  = (const int*)d_in[5];
  const int* lg_src = (const int*)d_in[6];
  const int* lg_dst = (const int*)d_in[7];
  const int* fg_src = (const int*)d_in[8];
  const int* fg_dst = (const int*)d_in[9];

  // --- workspace carving ---
  char* wsp = (char*)d_ws;
  auto balloc = [&](size_t bytes) -> void* {
    void* p = (void*)wsp; wsp += (bytes + 255) & ~(size_t)255; return p;
  };

  // --- parse params (jax pytree flatten order: dict keys sorted) ---
  int cur = 10;
  LinP lins[68]; int nlin = 0;
  auto plin = [&]() -> LinP {
    LinP L; L.b = (const float*)d_in[cur++]; L.w = (const float*)d_in[cur++];
    L.wt = (_Float16*)balloc((size_t)DD * DD * sizeof(_Float16));
    lins[nlin++] = L; return L;
  };
  auto pln = [&]() -> LnP {
    LnP L; L.b = (const float*)d_in[cur++]; L.g = (const float*)d_in[cur++]; return L;
  };
  auto peggc = [&]() -> EggcP {  // keys: A,B,C,Wd,Ws,ln_e,ln_n
    EggcP E; E.A = plin(); E.B = plin(); E.C = plin(); E.Wd = plin(); E.Ws = plin();
    E.ln_e = pln(); E.ln_n = pln(); return E;
  };
  EggcP al_edge[4], al_node[4], gnn[4];
  for (int i = 0; i < 4; ++i) { al_edge[i] = peggc(); al_node[i] = peggc(); }  // {edge,node}
  for (int i = 0; i < 4; ++i) gnn[i] = peggc();
  LinP lin1 = plin(), lin2 = plin();
  LnP  lnh  = pln();
  // mha keys sorted: e,k,ln_f,ln_x,o,oe,q,v
  LinP mE = plin(), mK = plin();
  LnP  mLNf = pln(), mLNx = pln();
  LinP mO = plin(), mOe = plin(), mQ = plin(), mV = plin();

  // --- big scratch ---
  const size_t NNmax = 240000, NEmax = 400000;
  float* P0 = (float*)balloc(NNmax * DD * 4);   // A / Q
  float* P1 = (float*)balloc(NNmax * DD * 4);   // B / K / O
  float* P2 = (float*)balloc(NNmax * DD * 4);   // Ws / V
  float* P3 = (float*)balloc(NNmax * DD * 4);   // Wd / h
  float* P4 = (float*)balloc(NEmax * DD * 4);   // Cy -> e  /  Ef -> S -> f_new
  float* P5 = (float*)balloc(NNmax * DD * 4);   // num
  float* P6 = (float*)balloc(NNmax * DD * 4);   // den
  float* xbuf = (float*)balloc((size_t)N * DD * 4);
  float*    Lg   = (float*)balloc((size_t)EF * 4 * 4);
  unsigned* Lmax = (unsigned*)balloc((size_t)N * 4 * 4);
  float*    mden = (float*)balloc((size_t)N * 4 * 4);

  float* out_x = (float*)d_out;
  float* out_y = out_x + (size_t)N * DD;
  float* out_z = out_y + (size_t)EG * DD;
  float* out_f = out_z + (size_t)T * DD;

  // --- convert all weights once per launch (idempotent) ---
  for (int i = 0; i < nlin; ++i)
    k_wconv<<<64, 256, 0, stream>>>(lins[i].w, lins[i].wt);

  hipMemcpyAsync(xbuf, in_x, (size_t)N * DD * 4, hipMemcpyDeviceToDevice, stream);
  hipMemcpyAsync(out_y, in_y, (size_t)EG * DD * 4, hipMemcpyDeviceToDevice, stream);
  hipMemcpyAsync(out_z, in_z, (size_t)T * DD * 4, hipMemcpyDeviceToDevice, stream);
  hipMemcpyAsync(out_f, in_f, (size_t)EF * DD * 4, hipMemcpyDeviceToDevice, stream);

  auto gemm = [&](const float* X, const LinP& L, float* Y, int M, int act) {
    dim3 g((unsigned)((M + 63) / 64));
    if (act) k_gemm<1><<<g, 256, 0, stream>>>(X, L.wt, L.b, Y, M);
    else     k_gemm<0><<<g, 256, 0, stream>>>(X, L.wt, L.b, Y, M);
  };

  // ================= MHA layer =================
  gemm(xbuf, mQ, P0, N, 0);
  gemm(xbuf, mK, P1, N, 0);
  gemm(xbuf, mV, P2, N, 0);
  gemm(out_f, mE, P4, EF, 0);                       // e-projection of f
  hipMemsetAsync(Lmax, 0, (size_t)N * 4 * 4, stream);
  hipMemsetAsync(mden, 0, (size_t)N * 4 * 4, stream);
  hipMemsetAsync(P5, 0, (size_t)N * DD * 4, stream);
  k_mha_edge1<<<(EF * 4 + 7) / 8, 256, 0, stream>>>(P0, P1, P4, Lg, Lmax,
                                                    fg_src, fg_dst, EF);
  k_mha_edge2<<<(unsigned)(((size_t)EF * DD + 255) / 256), 256, 0, stream>>>(
      P2, Lg, Lmax, P5, mden, fg_src, fg_dst, EF);
  k_mha_hcomb<<<(unsigned)(((size_t)N * DD + 255) / 256), 256, 0, stream>>>(
      P5, mden, P3, N);
  gemm(P3, mO, P1, N, 0);
  k_ln_add<<<(N + 7) / 8, 256, 0, stream>>>(xbuf, xbuf, P1, mLNx.g, mLNx.b, N);
  gemm(P4, mOe, P4, EF, 0);                          // f_new, in place (tile-safe)
  k_ln_add<<<(EF + 7) / 8, 256, 0, stream>>>(out_f, out_f, P4, mLNf.g, mLNf.b, EF);

  // ================= eggc layers =================
  auto run_eggc = [&](const EggcP& p, const int* src, const int* dst,
                      float* x_io, float* y_io, int NN, int NE) {
    gemm(x_io, p.A, P0, NN, 0);
    gemm(x_io, p.B, P1, NN, 0);
    gemm(x_io, p.Ws, P2, NN, 0);
    gemm(x_io, p.Wd, P3, NN, 0);
    gemm(y_io, p.C, P4, NE, 0);                      // Cy, becomes e in place
    hipMemsetAsync(P5, 0, (size_t)NN * DD * 4, stream);
    hipMemsetAsync(P6, 0, (size_t)NN * DD * 4, stream);
    k_eggc_edge<<<(unsigned)(((size_t)NE * DD + 255) / 256), 256, 0, stream>>>(
        P0, P1, P2, P4, P5, P6, src, dst, NE);
    k_eggc_node<<<(NN + 7) / 8, 256, 0, stream>>>(x_io, P3, P5, P6,
                                                  p.ln_n.g, p.ln_n.b, NN);
    k_ln_silu_add<<<(NE + 7) / 8, 256, 0, stream>>>(y_io, P4, p.ln_e.g, p.ln_e.b, NE);
  };

  for (int i = 0; i < 4; ++i) {                      // ALIGNN: node then edge
    run_eggc(al_node[i], g_src, g_dst, xbuf, out_y, N, EG);     // x,m (m=out_y)
    run_eggc(al_edge[i], lg_src, lg_dst, out_y, out_z, EG, T);  // y,z
  }
  for (int i = 0; i < 4; ++i)
    run_eggc(gnn[i], g_src, g_dst, xbuf, out_y, N, EG);

  // ================= head: out = LN(lin2(relu(lin1(x)))) + x =================
  gemm(xbuf, lin1, P0, N, 1);
  gemm(P0, lin2, P1, N, 0);
  k_ln_add<<<(N + 7) / 8, 256, 0, stream>>>(out_x, xbuf, P1, lnh.g, lnh.b, N);
}
